// WindowAttention_51883204935903
// MI455X (gfx1250) — matrix operand order repaired and checked by reading, not compiled, not run on hardware
//
#include <hip/hip_runtime.h>

// ---------------------------------------------------------------------------
// Swin window attention for MI455X (gfx1250): one workgroup (8 wave32) per
// 7x7 window, all GEMMs via v_wmma_f32_16x16x32_f16, fp32 accum.
// Weights are pre-converted to f16 in d_ws (one tiny pre-pass) so the GEMM
// inner loops issue pure b128 loads with no f32->f16 cvt chains.
// ---------------------------------------------------------------------------

typedef _Float16 v16h __attribute__((ext_vector_type(16)));
typedef _Float16 v8h  __attribute__((ext_vector_type(8)));
typedef _Float16 v4h  __attribute__((ext_vector_type(4)));
typedef float    v8f  __attribute__((ext_vector_type(8)));

#define SDIM   128
#define SNWIN  49
#define SH_IMG 112

#define QKVW_ELEMS  (384 * 128)
#define PROJW_ELEMS (128 * 128)
#define W16_BYTES   ((QKVW_ELEMS + PROJW_ELEMS) * 2)

static __device__ __forceinline__ v8f wmma_f16(v16h a, v16h b, v8f c) {
    return __builtin_amdgcn_wmma_f32_16x16x32_f16(false, a, false, b,
                                                  (short)0, c, false, false);
}

// A-tile (16x32 f16) from row-major f16 LDS. Contiguous K order:
// lanes 0-15 / 16-31 hold K=[ko..ko+15], ko = halfid*16.
static __device__ __forceinline__ v16h load_a16(const _Float16* base, int row,
                                                int stride, int koffs, int halfid) {
    const _Float16* p = base + row * stride + koffs + (halfid << 4);
    v8h lo = *(const v8h*)(p);
    v8h hh = *(const v8h*)(p + 8);
    return __builtin_shufflevector(lo, hh, 0,1,2,3,4,5,6,7,8,9,10,11,12,13,14,15);
}

// B-tile (32x16 f16): per lane fixed N=lane16, contiguous K run of 16 at
// koffs + halfid*16, read from f16 storage where rows are the N dimension.
static __device__ __forceinline__ v16h load_b16(const _Float16* base, int nrow,
                                                int stride, int koffs, int halfid) {
    const _Float16* p = base + nrow * stride + koffs + (halfid << 4);
    v8h lo = *(const v8h*)(p);
    v8h hh = *(const v8h*)(p + 8);
    return __builtin_shufflevector(lo, hh, 0,1,2,3,4,5,6,7,8,9,10,11,12,13,14,15);
}

// Fallback: B-tile from fp32 weight matrix, converted on the fly.
static __device__ __forceinline__ v16h load_bw_f32(const float* __restrict__ w,
                                                   int nglob, int koffs, int halfid) {
    const float* p = w + nglob * SDIM + koffs + (halfid << 4);
    const float4 f0 = *(const float4*)(p);
    const float4 f1 = *(const float4*)(p + 4);
    const float4 f2 = *(const float4*)(p + 8);
    const float4 f3 = *(const float4*)(p + 12);
    v16h r;
    r[0]=(_Float16)f0.x; r[1]=(_Float16)f0.y; r[2]=(_Float16)f0.z; r[3]=(_Float16)f0.w;
    r[4]=(_Float16)f1.x; r[5]=(_Float16)f1.y; r[6]=(_Float16)f1.z; r[7]=(_Float16)f1.w;
    r[8]=(_Float16)f2.x; r[9]=(_Float16)f2.y; r[10]=(_Float16)f2.z; r[11]=(_Float16)f2.w;
    r[12]=(_Float16)f3.x; r[13]=(_Float16)f3.y; r[14]=(_Float16)f3.z; r[15]=(_Float16)f3.w;
    return r;
}

// Shift-mask region id of a token in window (wy,wx) (original-image coords).
static __device__ __forceinline__ int region_of(int tok, int wy, int wx) {
    int ty = tok / 7, tx = tok - ty * 7;
    int y = wy * 7 + ty + 3; if (y >= SH_IMG) y -= SH_IMG;
    int x = wx * 7 + tx + 3; if (x >= SH_IMG) x -= SH_IMG;
    int rh = (y < 105) ? 0 : ((y < 109) ? 1 : 2);
    int rw = (x < 105) ? 0 : ((x < 109) ? 1 : 2);
    return rh * 3 + rw;
}

// Pre-pass: convert qkv_w + proj_w fp32 -> f16 into workspace (once per call;
// L2-resident thereafter, read by all 4096 attention blocks as f16).
__global__ __launch_bounds__(256)
void convert_weights(const float* __restrict__ qkvw,
                     const float* __restrict__ projw,
                     _Float16* __restrict__ w16) {
    int i = blockIdx.x * 256 + threadIdx.x;           // 65536 threads
    if (i < QKVW_ELEMS)                 w16[i] = (_Float16)qkvw[i];
    else if (i < QKVW_ELEMS + PROJW_ELEMS) w16[i] = (_Float16)projw[i - QKVW_ELEMS];
}

template <bool USE_W16>
__global__ __launch_bounds__(256)
void swin_window_attn(const float* __restrict__ xin,
                      const float* __restrict__ qkv_w,
                      const float* __restrict__ qkv_b,
                      const float* __restrict__ proj_w,
                      const float* __restrict__ proj_b,
                      const float* __restrict__ rpb,
                      const _Float16* __restrict__ w16,
                      float* __restrict__ out) {
    __shared__ __align__(256) unsigned char smem[65536];
    _Float16* sx  = (_Float16*)(smem);           // [64][128] staged window (f16)
    _Float16* sq  = (_Float16*)(smem + 16384);   // [64][128] Q (pre-scaled)
    _Float16* sk  = (_Float16*)(smem + 32768);   // [64][128] K
    _Float16* svT = (_Float16*)(smem + 49152);   // [128][64] V transposed
    _Float16* sp  = (_Float16*)(smem);           // [4][64][64] softmax(P), aliases sx+sq
    _Float16* sO  = (_Float16*)(smem + 32768);   // [64][128] attn output, aliases sk

    const _Float16* qkv_w16  = w16;               // [384][128]
    const _Float16* proj_w16 = w16 + QKVW_ELEMS;  // [128][128]

    const int tid    = threadIdx.x;
    const int wave   = tid >> 5;
    const int lane   = tid & 31;
    const int lane16 = lane & 15;
    const int halfid = lane >> 4;

    const int bb   = blockIdx.x >> 8;   // batch
    const int widx = blockIdx.x & 255;
    const int wy   = widx >> 4;
    const int wx   = widx & 15;

    // ---- Phase 0: stage cyclically-shifted 7x7 window -> LDS f16 (pad to 64 rows)
    for (int idx = tid; idx < 64 * 32; idx += 256) {
        int m  = idx >> 5;
        int cg = (idx & 31) << 2;
        if (m < SNWIN) {
            int ty = m / 7, tx = m - ty * 7;
            int y = wy * 7 + ty + 3; if (y >= SH_IMG) y -= SH_IMG;
            int x = wx * 7 + tx + 3; if (x >= SH_IMG) x -= SH_IMG;
            const float4 v = *(const float4*)(xin + ((((bb * SH_IMG) + y) * SH_IMG + x) << 7) + cg);
            v4h hv; hv[0]=(_Float16)v.x; hv[1]=(_Float16)v.y; hv[2]=(_Float16)v.z; hv[3]=(_Float16)v.w;
            *(v4h*)(sx + (m << 7) + cg) = hv;
        } else {
            v4h z = {};
            *(v4h*)(sx + (m << 7) + cg) = z;
        }
    }
    __syncthreads();

    // ---- Phase 1: fused QKV GEMM (64x128)@(128x384): 96 C-tiles over 8 waves
    const float SCALE = 0.17677669529663688f; // HEAD^-0.5
    for (int t = wave; t < 96; t += 8) {
        int mt = t & 3, nt = t >> 2;
        int ng = (nt << 4) + lane16;           // 0..383
        v8f c = {0.f,0.f,0.f,0.f,0.f,0.f,0.f,0.f};
        #pragma unroll
        for (int ks = 0; ks < 128; ks += 32) {
            v16h a  = load_a16(sx, (mt << 4) + lane16, SDIM, ks, halfid);
            v16h bm = USE_W16 ? load_b16(qkv_w16, ng, SDIM, ks, halfid)
                              : load_bw_f32(qkv_w, ng, ks, halfid);
            c = wmma_f16(a, bm, c);
        }
        float bias = qkv_b[ng];
        int mb = (mt << 4) + (halfid << 3);
        if (ng < 128) {                 // Q (scaled)
            #pragma unroll
            for (int e = 0; e < 8; ++e) sq[((mb + e) << 7) + ng] = (_Float16)((c[e] + bias) * SCALE);
        } else if (ng < 256) {          // K
            int nk = ng - 128;
            #pragma unroll
            for (int e = 0; e < 8; ++e) sk[((mb + e) << 7) + nk] = (_Float16)(c[e] + bias);
        } else {                        // V, stored transposed [feat][token]
            int nv = ng - 256;
            v8h hv;
            #pragma unroll
            for (int e = 0; e < 8; ++e) hv[e] = (_Float16)(c[e] + bias);
            *(v8h*)(svT + (nv << 6) + mb) = hv;
        }
    }
    __syncthreads();

    // ---- Phase 2: S = Q K^T per head, + rel-pos bias + shift mask, softmax.
    // wave w -> head w/2, row-tile pair (w&1)*2 + {0,1}. S held in registers.
    const int h   = wave >> 1;
    const int mt2 = (wave & 1) << 1;
    float sreg[2][4][8];

    int jh_[4], jw_[4], rj_[4]; bool jv_[4];
    #pragma unroll
    for (int nt = 0; nt < 4; ++nt) {
        int j = (nt << 4) + lane16;
        jv_[nt] = (j < SNWIN);
        int t7 = j / 7;
        jh_[nt] = t7; jw_[nt] = j - t7 * 7;
        rj_[nt] = jv_[nt] ? region_of(j, wy, wx) : -1;
    }

    #pragma unroll
    for (int mi = 0; mi < 2; ++mi) {
        int mt = mt2 + mi;
        v16h a = load_a16(sq, (mt << 4) + lane16, SDIM, h << 5, halfid);
        v8f cc[4];
        #pragma unroll
        for (int nt = 0; nt < 4; ++nt) {
            v16h bm = load_b16(sk, (nt << 4) + lane16, SDIM, h << 5, halfid);
            v8f z = {0.f,0.f,0.f,0.f,0.f,0.f,0.f,0.f};
            cc[nt] = wmma_f16(a, bm, z);
        }
        #pragma unroll
        for (int e = 0; e < 8; ++e) {
            int i = (mt << 4) + (halfid << 3) + e;
            bool iv = (i < SNWIN);
            int ih = 0, iw = 0, ri = 0;
            if (iv) { ih = i / 7; iw = i - ih * 7; ri = region_of(i, wy, wx); }
            #pragma unroll
            for (int nt = 0; nt < 4; ++nt) {
                float v;
                if (iv && jv_[nt]) {
                    int rel = (ih - jh_[nt] + 6) * 13 + (iw - jw_[nt] + 6);
                    v = cc[nt][e] + rpb[rel * 4 + h] + ((ri != rj_[nt]) ? -100.0f : 0.0f);
                } else {
                    v = -1e30f;
                }
                cc[nt][e] = v;
            }
            // row softmax: row lives in one VGPR slot across a 16-lane half
            float m = fmaxf(fmaxf(cc[0][e], cc[1][e]), fmaxf(cc[2][e], cc[3][e]));
            #pragma unroll
            for (int off = 8; off >= 1; off >>= 1) m = fmaxf(m, __shfl_xor(m, off, 16));
            float s = 0.f;
            #pragma unroll
            for (int nt = 0; nt < 4; ++nt) { float ev = __expf(cc[nt][e] - m); cc[nt][e] = ev; s += ev; }
            #pragma unroll
            for (int off = 8; off >= 1; off >>= 1) s += __shfl_xor(s, off, 16);
            float inv = 1.0f / s;
            #pragma unroll
            for (int nt = 0; nt < 4; ++nt) sreg[mi][nt][e] = cc[nt][e] * inv;
        }
    }
    __syncthreads();   // q/k reads done; P may overwrite x+q region
    #pragma unroll
    for (int mi = 0; mi < 2; ++mi) {
        int mt = mt2 + mi;
        #pragma unroll
        for (int nt = 0; nt < 4; ++nt)
            #pragma unroll
            for (int e = 0; e < 8; ++e)
                sp[((h << 6) + (mt << 4) + (halfid << 3) + e) * 64 + (nt << 4) + lane16] =
                    (_Float16)sreg[mi][nt][e];
    }
    __syncthreads();

    // ---- Phase 3: O = P V per head (P: 64x64, V: 64x32). 32 tiles over 8 waves.
    for (int t = wave; t < 32; t += 8) {
        int h2 = t >> 3, r = t & 7;
        int mt = r >> 1, nt2 = r & 1;
        v8f c = {0.f,0.f,0.f,0.f,0.f,0.f,0.f,0.f};
        #pragma unroll
        for (int ks = 0; ks < 64; ks += 32) {
            v16h a  = load_a16(sp + (h2 << 12), (mt << 4) + lane16, 64, ks, halfid);
            v16h bm = load_b16(svT, (h2 << 5) + (nt2 << 4) + lane16, 64, ks, halfid);
            c = wmma_f16(a, bm, c);
        }
        int fn = (h2 << 5) + (nt2 << 4) + lane16;  // head-major feature concat
        int mb = (mt << 4) + (halfid << 3);
        #pragma unroll
        for (int e = 0; e < 8; ++e) sO[((mb + e) << 7) + fn] = (_Float16)c[e];
    }
    __syncthreads();

    // ---- Phase 4: proj GEMM (64x128)@(128x128) + bias, reverse shift, store fp32
    for (int t = wave; t < 32; t += 8) {
        int mt = t & 3, nt = t >> 2;
        int ng = (nt << 4) + lane16;
        v8f c = {0.f,0.f,0.f,0.f,0.f,0.f,0.f,0.f};
        #pragma unroll
        for (int ks = 0; ks < 128; ks += 32) {
            v16h a  = load_a16(sO, (mt << 4) + lane16, SDIM, ks, halfid);
            v16h bm = USE_W16 ? load_b16(proj_w16, ng, SDIM, ks, halfid)
                              : load_bw_f32(proj_w, ng, ks, halfid);
            c = wmma_f16(a, bm, c);
        }
        float pb = proj_b[ng];
        int mb = (mt << 4) + (halfid << 3);
        #pragma unroll
        for (int e = 0; e < 8; ++e) {
            int i = mb + e;
            if (i < SNWIN) {
                int ty = i / 7, tx = i - ty * 7;
                int y = wy * 7 + ty + 3; if (y >= SH_IMG) y -= SH_IMG;
                int x = wx * 7 + tx + 3; if (x >= SH_IMG) x -= SH_IMG;
                out[((((bb * SH_IMG) + y) * SH_IMG + x) << 7) + ng] = c[e] + pb;
            }
        }
    }
}

extern "C" void kernel_launch(void* const* d_in, const int* in_sizes, int n_in,
                              void* d_out, int out_size, void* d_ws, size_t ws_size,
                              hipStream_t stream) {
    (void)in_sizes; (void)n_in; (void)out_size;
    const float* xin   = (const float*)d_in[0];
    const float* qkvw  = (const float*)d_in[1];
    const float* qkvb  = (const float*)d_in[2];
    const float* projw = (const float*)d_in[3];
    const float* projb = (const float*)d_in[4];
    const float* rpb   = (const float*)d_in[5];
    float* out = (float*)d_out;

    dim3 grid(4096);   // 16 batches * 256 windows
    dim3 block(256);   // 8 wave32

    const bool use_w16 = (d_ws != nullptr) && (ws_size >= (size_t)W16_BYTES);
    if (use_w16) {
        _Float16* w16 = (_Float16*)d_ws;
        hipLaunchKernelGGL(convert_weights, dim3(256), dim3(256), 0, stream,
                           qkvw, projw, w16);
        hipLaunchKernelGGL((swin_window_attn<true>), grid, block, 0, stream,
                           xin, qkvw, qkvb, projw, projb, rpb, w16, out);
    } else {
        hipLaunchKernelGGL((swin_window_attn<false>), grid, block, 0, stream,
                           xin, qkvw, qkvb, projw, projb, rpb,
                           (const _Float16*)nullptr, out);
    }
}
